// SupervisedGraphSage_27324581937608
// MI455X (gfx1250) — compile-verified, matrix operand order checked by reading, not run on hardware
//
#include <hip/hip_runtime.h>
#include <hip/hip_bf16.h>

// ---------------- CDNA5 (gfx1250) types ----------------
typedef __bf16 bf16_t;
typedef __attribute__((ext_vector_type(16))) __bf16 v16bf;
typedef __attribute__((ext_vector_type(4)))  __bf16 v4bf;
typedef __attribute__((ext_vector_type(8)))  float  v8f;

#define NNODES 100000
#define NEDGES 1600000
#define DF     128          // D_IN == HIDDEN == 128
#define NCLASS 40
#define NCPAD  48           // 40 padded to 3x16 WMMA column blocks
#define NTILES (NNODES / 16)   // 6250 exact

// ---------------- elementwise helpers ----------------
__global__ void cvt_f32_bf16_kernel(const float* __restrict__ in,
                                    bf16_t* __restrict__ out, int n4) {
  int i = blockIdx.x * blockDim.x + threadIdx.x;
  if (i >= n4) return;
  float4 v = ((const float4*)in)[i];
  v4bf o = {(bf16_t)v.x, (bf16_t)v.y, (bf16_t)v.z, (bf16_t)v.w};
  ((v4bf*)out)[i] = o;
}

__global__ void zero_f32_kernel(float* __restrict__ p, int n4) {
  int i = blockIdx.x * blockDim.x + threadIdx.x;
  if (i >= n4) return;
  float4 z = {0.f, 0.f, 0.f, 0.f};
  ((float4*)p)[i] = z;
}

// ---------------- edge scatter (mean-aggregate numerator + degree) ----------
// One wave32 per edge; each lane owns 4 consecutive features (32 * 4 = 128).
__global__ void scatter_f32_kernel(const float* __restrict__ x,
                                   const int* __restrict__ src,
                                   const int* __restrict__ dst,
                                   float* __restrict__ sum,
                                   float* __restrict__ deg, int nEdges) {
  int gtid = blockIdx.x * blockDim.x + threadIdx.x;
  int e = gtid >> 5, lane = gtid & 31;
  if (e >= nEdges) return;
  int s = src[e], d = dst[e];
  float4 v = *(const float4*)(x + (size_t)s * DF + lane * 4);
  float* sp = sum + (size_t)d * DF + lane * 4;
  atomicAdd(sp + 0, v.x); atomicAdd(sp + 1, v.y);
  atomicAdd(sp + 2, v.z); atomicAdd(sp + 3, v.w);
  if (lane == 0) atomicAdd(deg + d, 1.0f);
}

__global__ void scatter_bf16_kernel(const bf16_t* __restrict__ x,
                                    const int* __restrict__ src,
                                    const int* __restrict__ dst,
                                    float* __restrict__ sum, int nEdges) {
  int gtid = blockIdx.x * blockDim.x + threadIdx.x;
  int e = gtid >> 5, lane = gtid & 31;
  if (e >= nEdges) return;
  int s = src[e], d = dst[e];
  v4bf v = *(const v4bf*)(x + (size_t)s * DF + lane * 4);
  float* sp = sum + (size_t)d * DF + lane * 4;
  atomicAdd(sp + 0, (float)v.x); atomicAdd(sp + 1, (float)v.y);
  atomicAdd(sp + 2, (float)v.z); atomicAdd(sp + 3, (float)v.w);
}

// agg = bf16(sum / max(deg, 1))
__global__ void finalize_agg_kernel(const float* __restrict__ sum,
                                    const float* __restrict__ deg,
                                    bf16_t* __restrict__ agg, int n4) {
  int i = blockIdx.x * blockDim.x + threadIdx.x;
  if (i >= n4) return;
  int node = i >> 5;                       // 32 float4's per node (128 feats)
  float r = 1.0f / fmaxf(deg[node], 1.0f);
  float4 v = ((const float4*)sum)[i];
  v4bf o = {(bf16_t)(v.x * r), (bf16_t)(v.y * r),
            (bf16_t)(v.z * r), (bf16_t)(v.w * r)};
  ((v4bf*)agg)[i] = o;
}

// ---------------- WMMA GEMM layer 1: relu([xs|xa] @ W1) -> bf16 ------------
// 8 waves per block, wave w owns output columns [16w, 16w+16).
// K = 256 -> 8 chunks of 32 -> 8x v_wmma_f32_16x16x32_bf16 per 16x16 tile.
// All 8 A-fragments are issued as one load clause before the WMMA chain so
// load latency overlaps the matrix pipe; next tile's rows are prefetched.
// Per-lane fragment convention (same K-permutation for A and B, which the
// WMMA dot product is invariant to): lane = idx16 + 16*half, the 16 packed
// bf16 cover K = kc*32 + half*16 + [0,16) contiguously.
__global__ __launch_bounds__(256)
void gemm1_wmma_kernel(const bf16_t* __restrict__ xs,   // self  [N,128] bf16
                       const bf16_t* __restrict__ xa,   // agg   [N,128] bf16
                       const float*  __restrict__ W,    // [256,128] f32
                       bf16_t* __restrict__ out,        // [N,128] bf16 (relu'd)
                       int nTiles) {
  __shared__ bf16_t sW[256 * 128];                      // 64 KB of 320 KB WGP LDS
  for (int i = threadIdx.x; i < 256 * 128; i += 256) sW[i] = (bf16_t)W[i];
  __syncthreads();

  const int wave = threadIdx.x >> 5;
  const int lane = threadIdx.x & 31;
  const int n16  = lane & 15;
  const int half = lane >> 4;
  const int ncol = wave * 16 + n16;                     // 0..127

  // Hoist all B fragments into registers (8 chunks x 8 VGPRs).
  v16bf B[8];
#pragma unroll
  for (int kc = 0; kc < 8; ++kc) {
    v16bf b;
    int rowBase = kc * 32 + half * 16;
#pragma unroll
    for (int s = 0; s < 16; ++s) b[s] = sW[(rowBase + s) * 128 + ncol];
    B[kc] = b;
  }

  for (int tile = blockIdx.x; tile < nTiles; tile += gridDim.x) {
    const size_t row = (size_t)tile * 16 + n16;
    const bf16_t* ps = xs + row * DF + half * 16;
    const bf16_t* pa = xa + row * DF + half * 16;

    // Issue all 16 b128 loads up front (one clause), then consume.
    v16bf A[8];
#pragma unroll
    for (int kc = 0; kc < 4; ++kc) A[kc]     = *(const v16bf*)(ps + kc * 32);
#pragma unroll
    for (int kc = 0; kc < 4; ++kc) A[kc + 4] = *(const v16bf*)(pa + kc * 32);

    // Warm next grid-stride tile (global_prefetch_b8).
    int nt = tile + gridDim.x;
    if (nt < nTiles) {
      const size_t nrow = (size_t)nt * 16 + n16;
      __builtin_prefetch(xs + nrow * DF, 0, 0);
      __builtin_prefetch(xa + nrow * DF, 0, 0);
    }

    v8f c = {};
#pragma unroll
    for (int kc = 0; kc < 8; ++kc)
      c = __builtin_amdgcn_wmma_f32_16x16x32_bf16(false, A[kc], false, B[kc],
                                                  (short)0, c, false, false);

    // C/D layout: c[v] -> row (tile*16 + v + 8*half), col ncol.
    const size_t m0 = (size_t)tile * 16 + 8 * half;
#pragma unroll
    for (int v = 0; v < 8; ++v)
      out[(m0 + v) * DF + ncol] = (bf16_t)fmaxf(c[v], 0.0f);
  }
}

// ---------------- WMMA GEMM layer 2: [xs|xa] @ W2 -> f32 (padded 48) -------
__global__ __launch_bounds__(96)
void gemm2_wmma_kernel(const bf16_t* __restrict__ xs,
                       const bf16_t* __restrict__ xa,
                       const float*  __restrict__ W,    // [256,40] f32
                       float* __restrict__ out,         // [N,48] f32
                       int nTiles) {
  __shared__ bf16_t sW[256 * NCPAD];                    // 24 KB, zero-padded cols
  for (int i = threadIdx.x; i < 256 * NCPAD; i += 96) {
    int r = i / NCPAD, cidx = i % NCPAD;
    sW[i] = (cidx < NCLASS) ? (bf16_t)W[r * NCLASS + cidx] : (bf16_t)0.0f;
  }
  __syncthreads();

  const int wave = threadIdx.x >> 5;                    // 0..2 -> column block
  const int lane = threadIdx.x & 31;
  const int n16  = lane & 15;
  const int half = lane >> 4;
  const int ncol = wave * 16 + n16;                     // 0..47

  v16bf B[8];
#pragma unroll
  for (int kc = 0; kc < 8; ++kc) {
    v16bf b;
    int rowBase = kc * 32 + half * 16;
#pragma unroll
    for (int s = 0; s < 16; ++s) b[s] = sW[(rowBase + s) * NCPAD + ncol];
    B[kc] = b;
  }

  for (int tile = blockIdx.x; tile < nTiles; tile += gridDim.x) {
    const size_t row = (size_t)tile * 16 + n16;
    const bf16_t* ps = xs + row * DF + half * 16;
    const bf16_t* pa = xa + row * DF + half * 16;

    v16bf A[8];
#pragma unroll
    for (int kc = 0; kc < 4; ++kc) A[kc]     = *(const v16bf*)(ps + kc * 32);
#pragma unroll
    for (int kc = 0; kc < 4; ++kc) A[kc + 4] = *(const v16bf*)(pa + kc * 32);

    int nt = tile + gridDim.x;
    if (nt < nTiles) {
      const size_t nrow = (size_t)nt * 16 + n16;
      __builtin_prefetch(xs + nrow * DF, 0, 0);
      __builtin_prefetch(xa + nrow * DF, 0, 0);
    }

    v8f c = {};
#pragma unroll
    for (int kc = 0; kc < 8; ++kc)
      c = __builtin_amdgcn_wmma_f32_16x16x32_bf16(false, A[kc], false, B[kc],
                                                  (short)0, c, false, false);

    const size_t m0 = (size_t)tile * 16 + 8 * half;
#pragma unroll
    for (int v = 0; v < 8; ++v)
      out[(m0 + v) * NCPAD + ncol] = c[v];
  }
}

// ---------------- row-wise log_softmax over 40 classes ---------------------
// One wave per row; lane covers col=lane, and col=lane+32 when lane<8.
__global__ void logsoftmax_kernel(const float* __restrict__ enc,
                                  float* __restrict__ out, int nRows) {
  int gtid = blockIdx.x * blockDim.x + threadIdx.x;
  int row = gtid >> 5, lane = gtid & 31;
  if (row >= nRows) return;
  const float* p = enc + (size_t)row * NCPAD;
  float v0 = p[lane];
  float v1 = (lane < NCLASS - 32) ? p[lane + 32] : -__builtin_inff();
  float m = fmaxf(v0, v1);
#pragma unroll
  for (int off = 16; off; off >>= 1) m = fmaxf(m, __shfl_xor(m, off, 32));
  float s = expf(v0 - m) + ((lane < NCLASS - 32) ? expf(v1 - m) : 0.0f);
#pragma unroll
  for (int off = 16; off; off >>= 1) s += __shfl_xor(s, off, 32);
  float lse = m + logf(s);
  out[(size_t)row * NCLASS + lane] = v0 - lse;
  if (lane < NCLASS - 32) out[(size_t)row * NCLASS + 32 + lane] = v1 - lse;
}

// ---------------- driver ----------------
extern "C" void kernel_launch(void* const* d_in, const int* in_sizes, int n_in,
                              void* d_out, int out_size, void* d_ws, size_t ws_size,
                              hipStream_t stream) {
  const float* h   = (const float*)d_in[0];   // [100000,128]
  const float* W1  = (const float*)d_in[1];   // [256,128]
  const float* W2  = (const float*)d_in[2];   // [256,40]
  const int*   src = (const int*)d_in[3];     // [1.6M]
  const int*   dst = (const int*)d_in[4];     // [1.6M]
  float* out = (float*)d_out;                 // [100000,40]

  char* ws = (char*)d_ws;
  size_t o = 0;
  auto take = [&](size_t bytes) -> void* {
    void* p = ws + o;
    o += (bytes + 255) & ~(size_t)255;
    return p;
  };
  bf16_t* hb   = (bf16_t*)take((size_t)NNODES * DF * 2);   // bf16 of h
  bf16_t* x2   = (bf16_t*)take((size_t)NNODES * DF * 2);   // relu(enc1) bf16
  bf16_t* agg  = (bf16_t*)take((size_t)NNODES * DF * 2);   // mean-agg bf16
  float*  sum  = (float*) take((size_t)NNODES * DF * 4);   // scatter accumulator
  float*  deg  = (float*) take((size_t)NNODES * 4);        // degrees (reused)
  float*  enc2 = (float*) take((size_t)NNODES * NCPAD * 4);

  const int featN4 = NNODES * DF / 4;                      // 3.2M float4/v4bf
  const int edgeThreads = NEDGES * 32;

  // h -> bf16
  cvt_f32_bf16_kernel<<<(featN4 + 255) / 256, 256, 0, stream>>>(h, hb, featN4);

  // ---- layer 1 aggregation ----
  zero_f32_kernel<<<(featN4 + 255) / 256, 256, 0, stream>>>(sum, featN4);
  zero_f32_kernel<<<(NNODES / 4 + 255) / 256, 256, 0, stream>>>(deg, NNODES / 4);
  scatter_f32_kernel<<<(edgeThreads + 255) / 256, 256, 0, stream>>>(
      h, src, dst, sum, deg, NEDGES);
  finalize_agg_kernel<<<(featN4 + 255) / 256, 256, 0, stream>>>(sum, deg, agg, featN4);

  // enc1 = relu([h|agg] @ W1) -> bf16
  gemm1_wmma_kernel<<<1024, 256, 0, stream>>>(hb, agg, W1, x2, NTILES);

  // ---- layer 2 aggregation (degree reused) ----
  zero_f32_kernel<<<(featN4 + 255) / 256, 256, 0, stream>>>(sum, featN4);
  scatter_bf16_kernel<<<(edgeThreads + 255) / 256, 256, 0, stream>>>(
      x2, src, dst, sum, NEDGES);
  finalize_agg_kernel<<<(featN4 + 255) / 256, 256, 0, stream>>>(sum, deg, agg, featN4);

  // enc2 = [x2|agg] @ W2 (padded to 48 cols)
  gemm2_wmma_kernel<<<2048, 96, 0, stream>>>(x2, agg, W2, enc2, NTILES);

  // log_softmax rows
  logsoftmax_kernel<<<(NNODES * 32 + 255) / 256, 256, 0, stream>>>(enc2, out, NNODES);
}
